// LSTM_37847251812594
// MI455X (gfx1250) — compile-verified, matrix operand order checked by reading, not run on hardware
//
#include <hip/hip_runtime.h>

// ---------------------------------------------------------------------------
// 2-layer LSTM (B=256, T=64, 512->1024->1024) + classifier(1024->1000)+softmax
// bf16 WMMA path: V_WMMA_F32_16X16X32_BF16, f32 accumulate.
// ---------------------------------------------------------------------------

typedef __attribute__((ext_vector_type(16))) __bf16 v16bf;
typedef __attribute__((ext_vector_type(8)))  __bf16 v8bf;
typedef __attribute__((ext_vector_type(8)))  float  v8f;

constexpr int kB    = 256;    // batch
constexpr int kT    = 64;     // timesteps
constexpr int kNx0  = 512;    // layer0 input width
constexpr int kNh   = 1024;   // hidden width (both layers)
constexpr int kNv0  = 1536;   // kNh + kNx0
constexpr int kNv1  = 2048;   // kNh + kNh
constexpr int kNg   = 4096;   // 4 gates * kNh
constexpr int kCls  = 1000;   // classifier width

static __device__ __forceinline__ __bf16 f32_to_bf16(float f) {
  unsigned u = __builtin_bit_cast(unsigned, f);
  unsigned r = (u + 0x7fffu + ((u >> 16) & 1u)) >> 16;   // round-nearest-even
  unsigned short s = (unsigned short)r;
  return __builtin_bit_cast(__bf16, s);
}

// Load one 16x32 bf16 WMMA fragment row for this lane.
// ISA layout (16-bit A/B): lanes 0-15 hold K=k..k+7 (VGPR0-3) and K=k+16..k+23
// (VGPR4-7); lanes 16-31 hold K=k+8..k+15 and K=k+24..k+31. With p already
// offset by (lane>>4)*8, that is two contiguous 16-byte loads at +0 and +16.
static __device__ __forceinline__ v16bf ld_frag(const __bf16* p) {
  v8bf lo = *(const v8bf*)(p);
  v8bf hi = *(const v8bf*)(p + 16);
  return __builtin_shufflevector(lo, hi,
      0, 1, 2, 3, 4, 5, 6, 7, 8, 9, 10, 11, 12, 13, 14, 15);
}

// ---------------------------------------------------------------------------
// Fused 4-gate GEMM:  out[256, 4096] = concat(Ah[256,1024], Ax[256,Kx]) @ Wt^T
// Wt packed transposed: [4096 rows][n_v cols] bf16, row r = gate*1024 + n.
// Block = 256 threads = 8 waves; wave tile = 16(M) x 64(N) (4 wmma accums).
// Block tile = 32(M) x 256(N); grid = (4096/256, 256/32) = (16, 8).
// ---------------------------------------------------------------------------
__global__ __launch_bounds__(256)
void lstm_gates_gemm(const __bf16* __restrict__ Ah,
                     const __bf16* __restrict__ Ax, int ldx, int Kx,
                     const __bf16* __restrict__ Wt, int n_v,
                     float* __restrict__ out) {
  const int lane  = threadIdx.x & 31;
  const int wave  = threadIdx.x >> 5;
  const int tileM = blockIdx.y * 32 + (wave & 1) * 16;
  const int tileN = blockIdx.x * 256 + (wave >> 1) * 64;
  const int row   = lane & 15;
  const int kgrp  = (lane >> 4) << 3;     // 0 or 8

  v8f acc0 = {}, acc1 = {}, acc2 = {}, acc3 = {};

  const __bf16* a_h = Ah + (size_t)(tileM + row) * kNh + kgrp;
  const __bf16* b0  = Wt + (size_t)(tileN + row +  0) * n_v + kgrp;
  const __bf16* b1  = Wt + (size_t)(tileN + row + 16) * n_v + kgrp;
  const __bf16* b2  = Wt + (size_t)(tileN + row + 32) * n_v + kgrp;
  const __bf16* b3  = Wt + (size_t)(tileN + row + 48) * n_v + kgrp;

  // ---- recurrent (h) part of the concat: K = 0..1023 ----
  #pragma unroll 2
  for (int k = 0; k < kNh; k += 32) {
    __builtin_prefetch(b0 + k + 512, 0, 0);      // global_prefetch_b8
    v16bf a  = ld_frag(a_h + k);
    v16bf w0 = ld_frag(b0 + k);
    v16bf w1 = ld_frag(b1 + k);
    v16bf w2 = ld_frag(b2 + k);
    v16bf w3 = ld_frag(b3 + k);
    acc0 = __builtin_amdgcn_wmma_f32_16x16x32_bf16(false, a, false, w0, (short)0, acc0, false, false);
    acc1 = __builtin_amdgcn_wmma_f32_16x16x32_bf16(false, a, false, w1, (short)0, acc1, false, false);
    acc2 = __builtin_amdgcn_wmma_f32_16x16x32_bf16(false, a, false, w2, (short)0, acc2, false, false);
    acc3 = __builtin_amdgcn_wmma_f32_16x16x32_bf16(false, a, false, w3, (short)0, acc3, false, false);
  }

  // ---- input (x) part of the concat: K = 1024..1024+Kx-1 ----
  const __bf16* a_x = Ax + (size_t)(tileM + row) * ldx + kgrp;
  #pragma unroll 2
  for (int k = 0; k < Kx; k += 32) {
    v16bf a  = ld_frag(a_x + k);
    v16bf w0 = ld_frag(b0 + kNh + k);
    v16bf w1 = ld_frag(b1 + kNh + k);
    v16bf w2 = ld_frag(b2 + kNh + k);
    v16bf w3 = ld_frag(b3 + kNh + k);
    acc0 = __builtin_amdgcn_wmma_f32_16x16x32_bf16(false, a, false, w0, (short)0, acc0, false, false);
    acc1 = __builtin_amdgcn_wmma_f32_16x16x32_bf16(false, a, false, w1, (short)0, acc1, false, false);
    acc2 = __builtin_amdgcn_wmma_f32_16x16x32_bf16(false, a, false, w2, (short)0, acc2, false, false);
    acc3 = __builtin_amdgcn_wmma_f32_16x16x32_bf16(false, a, false, w3, (short)0, acc3, false, false);
  }

  // C/D layout: lanes 0-15 -> N=lane, M=vgpr; lanes 16-31 -> M=vgpr+8.
  const int mbase = (lane >> 4) << 3;
  const int ncol  = lane & 15;
  float* op = out + (size_t)(tileM + mbase) * kNg + tileN + ncol;
  #pragma unroll
  for (int v = 0; v < 8; ++v) {
    op[(size_t)v * kNg +  0] = acc0[v];
    op[(size_t)v * kNg + 16] = acc1[v];
    op[(size_t)v * kNg + 32] = acc2[v];
    op[(size_t)v * kNg + 48] = acc3[v];
  }
}

// ---------------------------------------------------------------------------
// Elementwise LSTM cell: gates [B,4096] (F,I,C,O major) -> c (f32), h (bf16[,f32])
// ---------------------------------------------------------------------------
__global__ __launch_bounds__(256)
void lstm_cell(const float* __restrict__ gates,
               const float* __restrict__ bf, const float* __restrict__ bi,
               const float* __restrict__ bc, const float* __restrict__ bo,
               float* __restrict__ c, __bf16* __restrict__ hbf,
               float* __restrict__ hf32) {
  int idx = blockIdx.x * blockDim.x + threadIdx.x;   // 256*1024
  int b = idx >> 10, j = idx & 1023;
  size_t g = (size_t)b * kNg + j;
  float fg = 1.f / (1.f + expf(-(gates[g        ] + bf[j])));
  float ig = 1.f / (1.f + expf(-(gates[g + 1024 ] + bi[j])));
  float cg = tanhf(gates[g + 2048] + bc[j]);
  float og = 1.f / (1.f + expf(-(gates[g + 3072 ] + bo[j])));
  float cn = fg * c[idx] + ig * cg;
  c[idx] = cn;
  float h = og * tanhf(cn);
  hbf[idx] = f32_to_bf16(h);
  if (hf32) hf32[idx] = h;
}

// x [B, T*512] f32  ->  xbf [T, B, 512] bf16
__global__ __launch_bounds__(256)
void cvt_x(const float* __restrict__ x, __bf16* __restrict__ xbf) {
  int idx = blockIdx.x * blockDim.x + threadIdx.x;   // T*B*512
  int t = idx >> 17;                // B*512 = 131072 = 2^17
  int r = idx & 131071;
  int b = r >> 9, k = r & 511;
  xbf[idx] = f32_to_bf16(x[(size_t)b * (kT * kNx0) + t * kNx0 + k]);
}

// src [n_v, 1024] f32 (k-major rows)  ->  dst [1024, n_v] bf16 (transposed)
__global__ __launch_bounds__(256)
void pack_wt(const float* __restrict__ src, __bf16* __restrict__ dst, int n_v) {
  long long idx = (long long)blockIdx.x * blockDim.x + threadIdx.x;
  if (idx >= (long long)n_v * kNh) return;
  int n = (int)(idx / n_v), k = (int)(idx % n_v);
  dst[idx] = f32_to_bf16(src[(size_t)k * kNh + n]);
}

// logits[b,n] = sum_k h[b,k] * W[k,n] + bias[n]   (f32 for final precision)
__global__ __launch_bounds__(256)
void classifier(const float* __restrict__ h, const float* __restrict__ W,
                const float* __restrict__ bias, float* __restrict__ out) {
  int idx = blockIdx.x * blockDim.x + threadIdx.x;   // 256*1000
  if (idx >= kB * kCls) return;
  int b = idx / kCls, n = idx - b * kCls;
  const float* hr = h + (size_t)b * kNh;
  float s = bias[n];
  #pragma unroll 4
  for (int k = 0; k < kNh; ++k) s += hr[k] * W[(size_t)k * kCls + n];
  out[idx] = s;
}

__global__ __launch_bounds__(256)
void softmax_rows(const float* __restrict__ logits, float* __restrict__ pred) {
  __shared__ float sh[256];
  int b = blockIdx.x, t = threadIdx.x;
  const float* row = logits + (size_t)b * kCls;
  float m = -1e30f;
  for (int i = t; i < kCls; i += 256) m = fmaxf(m, row[i]);
  sh[t] = m; __syncthreads();
  for (int s = 128; s > 0; s >>= 1) { if (t < s) sh[t] = fmaxf(sh[t], sh[t + s]); __syncthreads(); }
  m = sh[0]; __syncthreads();
  float sum = 0.f;
  for (int i = t; i < kCls; i += 256) sum += expf(row[i] - m);
  sh[t] = sum; __syncthreads();
  for (int s = 128; s > 0; s >>= 1) { if (t < s) sh[t] += sh[t + s]; __syncthreads(); }
  float inv = 1.f / sh[0];
  float* pr = pred + (size_t)b * kCls;
  for (int i = t; i < kCls; i += 256) pr[i] = expf(row[i] - m) * inv;
}

extern "C" void kernel_launch(void* const* d_in, const int* in_sizes, int n_in,
                              void* d_out, int out_size, void* d_ws, size_t ws_size,
                              hipStream_t stream) {
  (void)in_sizes; (void)n_in; (void)out_size; (void)ws_size;
  const float* x = (const float*)d_in[0];
  // d_in: 1..4 = F0,I0,C0,O0 ; 5..8 = bf0,bi0,bc0,bo0 ; 9..12 = F1..O1 ;
  //       13..16 = bf1..bo1 ; 17 = W ; 18 = b
  char* ws = (char*)d_ws;
  size_t o = 0;
  const size_t XBF = o; o += (size_t)kT * kB * kNx0 * 2;   // x in bf16
  const size_t W0  = o; o += (size_t)4 * kNh * kNv0 * 2;   // layer0 packed wt
  const size_t W1  = o; o += (size_t)4 * kNh * kNv1 * 2;   // layer1 packed wt
  const size_t H0  = o; o += (size_t)kB * kNh * 2;         // h0 bf16
  const size_t H1  = o; o += (size_t)kB * kNh * 2;         // h1 bf16
  const size_t C0  = o; o += (size_t)kB * kNh * 4;         // c0 f32
  const size_t C1  = o; o += (size_t)kB * kNh * 4;         // c1 f32
  const size_t H1F = o; o += (size_t)kB * kNh * 4;         // h1 f32 (for classifier)
  const size_t G   = o; o += (size_t)kB * kNg * 4;         // fused gate buffer

  // zero-init recurrent state (ws is poisoned; must init every call)
  hipMemsetAsync(ws + H0, 0, (size_t)kB * kNh * 2, stream);
  hipMemsetAsync(ws + H1, 0, (size_t)kB * kNh * 2, stream);
  hipMemsetAsync(ws + C0, 0, (size_t)kB * kNh * 4, stream);
  hipMemsetAsync(ws + C1, 0, (size_t)kB * kNh * 4, stream);

  // precision prep: activations & weights -> bf16 (weights also transposed)
  cvt_x<<<(kT * kB * kNx0) / 256, 256, 0, stream>>>(x, (__bf16*)(ws + XBF));
  for (int g = 0; g < 4; ++g) {
    pack_wt<<<(kNv0 * kNh + 255) / 256, 256, 0, stream>>>(
        (const float*)d_in[1 + g], (__bf16*)(ws + W0) + (size_t)g * kNh * kNv0, kNv0);
    pack_wt<<<(kNv1 * kNh + 255) / 256, 256, 0, stream>>>(
        (const float*)d_in[9 + g], (__bf16*)(ws + W1) + (size_t)g * kNh * kNv1, kNv1);
  }

  dim3 ggrid(kNg / 256, kB / 32);   // (16, 8)
  for (int t = 0; t < kT; ++t) {
    // layer 0: concat(h0, x_t) @ W0  -> gates -> (c0, h0)
    lstm_gates_gemm<<<ggrid, 256, 0, stream>>>(
        (const __bf16*)(ws + H0),
        (const __bf16*)(ws + XBF) + (size_t)t * kB * kNx0, kNx0, kNx0,
        (const __bf16*)(ws + W0), kNv0, (float*)(ws + G));
    lstm_cell<<<(kB * kNh) / 256, 256, 0, stream>>>(
        (const float*)(ws + G), (const float*)d_in[5], (const float*)d_in[6],
        (const float*)d_in[7], (const float*)d_in[8],
        (float*)(ws + C0), (__bf16*)(ws + H0), nullptr);
    // layer 1: concat(h1, h0) @ W1  -> gates -> (c1, h1)
    lstm_gates_gemm<<<ggrid, 256, 0, stream>>>(
        (const __bf16*)(ws + H1),
        (const __bf16*)(ws + H0), kNh, kNh,
        (const __bf16*)(ws + W1), kNv1, (float*)(ws + G));
    lstm_cell<<<(kB * kNh) / 256, 256, 0, stream>>>(
        (const float*)(ws + G), (const float*)d_in[13], (const float*)d_in[14],
        (const float*)d_in[15], (const float*)d_in[16],
        (float*)(ws + C1), (__bf16*)(ws + H1), (float*)(ws + H1F));
  }

  float* logits = (float*)d_out;
  float* pred   = (float*)d_out + (size_t)kB * kCls;
  classifier<<<(kB * kCls + 255) / 256, 256, 0, stream>>>(
      (const float*)(ws + H1F), (const float*)d_in[17], (const float*)d_in[18], logits);
  softmax_rows<<<kB, 256, 0, stream>>>(logits, pred);
}
